// HGTBaseline_59682865545615
// MI455X (gfx1250) — matrix operand order repaired and checked by reading, not compiled, not run on hardware
//
#include <hip/hip_runtime.h>
#include <hip/hip_bf16.h>
#include <float.h>

// ---------------------------------------------------------------------------
// HGT forward for MI455X (gfx1250), wave32, WMMA bf16 path, bf16-resident
// activations (memory-bound workload: halve HBM traffic vs fp32).
//
// Dense GEMMs run v_wmma_f32_16x16x32_bf16 with the full K panel staged in
// LDS (K is compile-time 64/96/128 -> fully unrolled WMMA stretch, one
// barrier pair per block).  Segment softmax / scatter aggregation stays on
// VALU + fp32 global atomics.
// ---------------------------------------------------------------------------

typedef __attribute__((ext_vector_type(8)))  __bf16 v8bf;
typedef __attribute__((ext_vector_type(16))) __bf16 v16bf;
typedef __attribute__((ext_vector_type(8)))  float  v8f;
typedef unsigned short ushort_t;

#define NT_EVENT   150000
#define NT_ENTITY  150000
#define NTOT_N     300000
#define HID        128
#define HEADS      4
#define DHEAD      32

__device__ inline ushort_t f2bf(float f) {
    unsigned u = __float_as_uint(f);
    unsigned r = u + 0x7fffu + ((u >> 16) & 1u);   // round-to-nearest-even
    return (ushort_t)(r >> 16);
}
__device__ inline float bf2f(ushort_t u) {
    return __uint_as_float(((unsigned)u) << 16);
}
__device__ inline unsigned floatFlip(float f) {     // monotonic float->uint
    unsigned u = __float_as_uint(f);
    return (u & 0x80000000u) ? ~u : (u | 0x80000000u);
}
__device__ inline float floatUnflip(unsigned u) {
    unsigned b = (u & 0x80000000u) ? (u ^ 0x80000000u) : ~u;
    return __uint_as_float(b);
}
__device__ inline v16bf cat8(v8bf lo, v8bf hi) {
    v16bf r;
#pragma unroll
    for (int i = 0; i < 8; ++i) { r[i] = lo[i]; r[i + 8] = hi[i]; }
    return r;
}

// ---------------------------------------------------------------------------
// Tiled GEMM:  Y[N,M] = X[N,K] @ W[K,M] + bias
//   X: fp32 or bf16 (xIsBf), W/bias: fp32, Y: fp32 or bf16 (yIsBf)
// Block: 256 threads = 8 waves; tile 128 rows x up-to-128 cols; full-K LDS
// panel; wave w owns column tile w, 8 row tiles -> 8 v8f accumulators,
// K/32 * 8 WMMAs back-to-back per wave between a single barrier pair.
// ---------------------------------------------------------------------------
template <int K>
__global__ __launch_bounds__(256)
void gemm_wmma(const void* __restrict__ Xv, int xIsBf,
               const float* __restrict__ W, const float* __restrict__ bias,
               void* __restrict__ Yv, int yIsBf, int Nrows, int M)
{
    __shared__ ushort_t ldsA[128 * K];    // A panel  [row][k]     (bf16)
    __shared__ ushort_t ldsBt[128 * K];   // B panel  [col][k]     (bf16, transposed)

    const int tid  = threadIdx.x;
    const int lane = tid & 31;
    const int wave = tid >> 5;
    const int rowBase   = blockIdx.x * 128;
    const int nColTiles = (M + 15) >> 4;

    // L2 warm for this block's A panel (global_prefetch_b8 on gfx1250)
    __builtin_prefetch((const char*)Xv + (size_t)rowBase * K * (xIsBf ? 2 : 4) +
                       (size_t)tid * 64, 0, 1);

    // stage A: 128 x K -> bf16 LDS
    if (xIsBf) {
        const ushort_t* X = (const ushort_t*)Xv;
#pragma unroll
        for (int i = 0; i < K / 2; ++i) {
            int e = i * 256 + tid;
            int r = e / K, c = e - r * K;
            int gr = rowBase + r;
            ldsA[e] = (gr < Nrows) ? X[(size_t)gr * K + c] : (ushort_t)0;
        }
    } else {
        const float* X = (const float*)Xv;
#pragma unroll
        for (int i = 0; i < K / 2; ++i) {
            int e = i * 256 + tid;
            int r = e / K, c = e - r * K;
            int gr = rowBase + r;
            ldsA[e] = (gr < Nrows) ? f2bf(X[(size_t)gr * K + c]) : (ushort_t)0;
        }
    }
    // stage B transposed: W[k][m] (fp32) -> ldsBt[m][k] (bf16)
#pragma unroll
    for (int i = 0; i < K / 2; ++i) {
        int e  = i * 256 + tid;
        int kk = e >> 7, m = e & 127;
        float v = (m < M) ? W[(size_t)kk * M + m] : 0.0f;
        ldsBt[m * K + kk] = f2bf(v);
    }
    __syncthreads();

    v8f acc[8];
#pragma unroll
    for (int r = 0; r < 8; ++r)
#pragma unroll
        for (int j = 0; j < 8; ++j) acc[r][j] = 0.0f;

    if (wave < nColTiles) {
        const int col  = (wave << 4) + (lane & 15);
        const int half = lane >> 4;
#pragma unroll
        for (int k0 = 0; k0 < K; k0 += 32) {
            // B fragment: K k0+{0..7},{16..23} (+8 if hi half), matching A
            const __bf16* bp = (const __bf16*)&ldsBt[col * K + k0 + (half << 3)];
            v16bf bfrag = cat8(*(const v8bf*)(bp), *(const v8bf*)(bp + 16));
#pragma unroll
            for (int r = 0; r < 8; ++r) {
                int ar = (r << 4) + (lane & 15);
                const __bf16* ap = (const __bf16*)&ldsA[ar * K + k0 + (half << 3)];
                v8bf lo = *(const v8bf*)(ap);         // K k0+{0..7}  (+8 if hi half)
                v8bf hi = *(const v8bf*)(ap + 16);    // K k0+{16..23}(+8 if hi half)
                v16bf afrag = cat8(lo, hi);
                acc[r] = __builtin_amdgcn_wmma_f32_16x16x32_bf16(
                    false, afrag, false, bfrag, (short)0, acc[r], false, false);
            }
        }

        const float bv = (col < M) ? bias[col] : 0.0f;
        const int rh = half << 3;                      // C/D: lanes 16-31 hold M+8
#pragma unroll
        for (int r = 0; r < 8; ++r) {
#pragma unroll
            for (int j = 0; j < 8; ++j) {
                int row = rowBase + (r << 4) + rh + j;
                if (row < Nrows && col < M) {
                    float v = acc[r][j] + bv;
                    if (yIsBf) ((ushort_t*)Yv)[(size_t)row * M + col] = f2bf(v);
                    else       ((float*)Yv)[(size_t)row * M + col] = v;
                }
            }
        }
    }
}

// ---------------------------------------------------------------------------
// Edge pass 1: a[e,h] = (q_dst . (k_src @ a_rel[h])) * p[h]/sqrt(D); seg-max.
// ---------------------------------------------------------------------------
__global__ __launch_bounds__(256)
void edge_attn(const int* __restrict__ ei, int E, int srcOff, int dstOff,
               const ushort_t* __restrict__ kbuf, const ushort_t* __restrict__ qbuf,
               const float* __restrict__ arel, const float* __restrict__ prel,
               float* __restrict__ aout, unsigned* __restrict__ amaxKey)
{
    __shared__ float sA[HEADS * DHEAD * DHEAD];
    for (int i = threadIdx.x; i < HEADS * DHEAD * DHEAD; i += 256) sA[i] = arel[i];
    __syncthreads();

    int t = blockIdx.x * 256 + threadIdx.x;
    int e = t >> 2, h = t & 3;
    if (e >= E) return;

    int src = ei[e] + srcOff;
    int dst = ei[E + e] + dstOff;
    const ushort_t* kp = kbuf + (size_t)src * HID + h * DHEAD;
    const ushort_t* qp = qbuf + (size_t)dst * HID + h * DHEAD;
    const float* A = sA + h * DHEAD * DHEAD;

    float m[DHEAD];
#pragma unroll
    for (int f = 0; f < DHEAD; ++f) m[f] = 0.0f;
    for (int d = 0; d < DHEAD; ++d) {
        float kd = bf2f(kp[d]);
#pragma unroll
        for (int f = 0; f < DHEAD; ++f) m[f] += kd * A[d * DHEAD + f];
    }
    float dot = 0.0f;
#pragma unroll
    for (int f = 0; f < DHEAD; ++f) dot += bf2f(qp[f]) * m[f];

    float val = dot * prel[h] * 0.17677669529663687f;  // 1/sqrt(32)
    aout[(size_t)e * HEADS + h] = val;
    atomicMax(&amaxKey[(size_t)dst * HEADS + h], floatFlip(val));
}

// Edge pass 2: ex = exp(a - max); accumulate denominator.
__global__ __launch_bounds__(256)
void edge_exp(const int* __restrict__ ei, int E, int dstOff,
              float* __restrict__ aout, const unsigned* __restrict__ amaxKey,
              float* __restrict__ den)
{
    int t = blockIdx.x * 256 + threadIdx.x;
    int e = t >> 2, h = t & 3;
    if (e >= E) return;
    int dst = ei[E + e] + dstOff;
    float gm = floatUnflip(amaxKey[(size_t)dst * HEADS + h]);
    float ex = __expf(aout[(size_t)e * HEADS + h] - gm);
    aout[(size_t)e * HEADS + h] = ex;
    atomicAdd(&den[(size_t)dst * HEADS + h], ex);
}

// Edge pass 3: vs = v_src @ m_rel[h]; scatter (a/den) * vs into fp32 agg.
__global__ __launch_bounds__(256)
void edge_agg(const int* __restrict__ ei, int E, int srcOff, int dstOff,
              const ushort_t* __restrict__ vbuf, const float* __restrict__ mrel,
              const float* __restrict__ aout, const float* __restrict__ den,
              float* __restrict__ agg)
{
    __shared__ float sM[HEADS * DHEAD * DHEAD];
    for (int i = threadIdx.x; i < HEADS * DHEAD * DHEAD; i += 256) sM[i] = mrel[i];
    __syncthreads();

    int t = blockIdx.x * 256 + threadIdx.x;
    int e = t >> 2, h = t & 3;
    if (e >= E) return;

    int src = ei[e] + srcOff;
    int dst = ei[E + e] + dstOff;
    float w = aout[(size_t)e * HEADS + h] / den[(size_t)dst * HEADS + h];

    const ushort_t* vp = vbuf + (size_t)src * HID + h * DHEAD;
    const float* M = sM + h * DHEAD * DHEAD;
    float vs[DHEAD];
#pragma unroll
    for (int f = 0; f < DHEAD; ++f) vs[f] = 0.0f;
    for (int d = 0; d < DHEAD; ++d) {
        float vd = bf2f(vp[d]);
#pragma unroll
        for (int f = 0; f < DHEAD; ++f) vs[f] += vd * M[d * DHEAD + f];
    }
    float* dstp = agg + (size_t)dst * HID + h * DHEAD;
#pragma unroll
    for (int f = 0; f < DHEAD; ++f) atomicAdd(&dstp[f], w * vs[f]);
}

// Elementwise helpers
__global__ __launch_bounds__(256)
void init_amax(unsigned* __restrict__ p, size_t n)
{
    size_t i = (size_t)blockIdx.x * 256 + threadIdx.x;
    if (i < n) p[i] = floatFlip(-FLT_MAX);
}

__global__ __launch_bounds__(256)
void gelu_to_bf(const float* __restrict__ x, ushort_t* __restrict__ y, size_t n)
{
    size_t i = (size_t)blockIdx.x * 256 + threadIdx.x;
    if (i < n) {
        float v = x[i];
        y[i] = f2bf(0.5f * v * (1.0f + erff(v * 0.70710678118654752f)));
    }
}

__global__ __launch_bounds__(256)
void residual_relu_bf(ushort_t* __restrict__ h, const ushort_t* __restrict__ hnew,
                      const float* __restrict__ skip, size_t n)
{
    float beta = 1.0f / (1.0f + __expf(-skip[0]));
    size_t i = (size_t)blockIdx.x * 256 + threadIdx.x;
    if (i < n) {
        float o = beta * bf2f(hnew[i]) + (1.0f - beta) * bf2f(h[i]);
        h[i] = f2bf(fmaxf(o, 0.0f));
    }
}

// ---------------------------------------------------------------------------
// Host orchestration.  Param leaves follow JAX pytree flattening
// (dict keys alphabetical, tuples in order):
//   layers[l]: a_rel{co,mentions,next} k{entity(W,b),event(W,b)} m_rel{..}
//              out_lin{..} p_rel{..} q{..} skip{entity,event} v{..}   (27 leaves)
//   lin: entity(W,b), event(W,b)      out: entity(W,b), event(W,b)
// ---------------------------------------------------------------------------
extern "C" void kernel_launch(void* const* d_in, const int* in_sizes, int n_in,
                              void* d_out, int out_size, void* d_ws, size_t ws_size,
                              hipStream_t stream)
{
    (void)in_sizes; (void)n_in; (void)out_size; (void)ws_size;

    const float* x_event  = (const float*)d_in[0];
    const float* x_entity = (const float*)d_in[1];
    const int*   ei_ment  = (const int*)d_in[2];
    const int*   ei_next  = (const int*)d_in[3];
    const int*   ei_co    = (const int*)d_in[4];
    const int P = 5;
    auto f = [&](int i) { return (const float*)d_in[i]; };

    const int    offE = NT_EVENT;
    const size_t nH   = (size_t)NTOT_N * HID;   // 38.4M elements

    // workspace: fp32 agg | bf16 h,k,q,v | fp32 edge logits | tables
    char* base = (char*)d_ws;
    float*    agg  = (float*)base;                          // nH fp32
    ushort_t* h    = (ushort_t*)(base + nH * 4);            // nH bf16
    ushort_t* kbuf = h + nH;
    ushort_t* qbuf = kbuf + nH;
    ushort_t* vbuf = qbuf + nH;
    float*    aE   = (float*)(base + nH * 4 + nH * 8);      // 2.2M fp32
    unsigned* amax = (unsigned*)(aE + 2200000);             // 1.2M
    float*    den  = (float*)(amax + 1200000);              // 1.2M
    ushort_t* hnew  = kbuf;   // alias: k dead after aggregation
    ushort_t* aggbf = qbuf;   // alias: q dead after edge passes

    auto gemm = [&](const void* X, int xbf, const float* W, const float* b,
                    void* Y, int ybf, int N, int K, int M) {
        dim3 g((N + 127) / 128), blk(256);
        if (K == 128)     gemm_wmma<128><<<g, blk, 0, stream>>>(X, xbf, W, b, Y, ybf, N, M);
        else if (K == 96) gemm_wmma<96> <<<g, blk, 0, stream>>>(X, xbf, W, b, Y, ybf, N, M);
        else              gemm_wmma<64> <<<g, blk, 0, stream>>>(X, xbf, W, b, Y, ybf, N, M);
    };
    auto eblocks = [](int E) { return dim3((E * 4 + 255) / 256); };

    // Input projections (fp32 in -> bf16 hidden)
    gemm(x_event,  0, f(P + 56), f(P + 57), h,        1, NT_EVENT,  96, HID);
    gemm(x_entity, 0, f(P + 54), f(P + 55), h + (size_t)offE * HID, 1, NT_ENTITY, 64, HID);

    for (int l = 0; l < 2; ++l) {
        const int B = P + l * 27;
        ushort_t* hEnt = h + (size_t)offE * HID;

        // k/q/v projections (bf16 WMMA GEMMs, bf16 out)
        gemm(h,    1, f(B + 5),  f(B + 6),  kbuf,                      1, NT_EVENT,  HID, HID);
        gemm(hEnt, 1, f(B + 3),  f(B + 4),  kbuf + (size_t)offE * HID, 1, NT_ENTITY, HID, HID);
        gemm(h,    1, f(B + 19), f(B + 20), qbuf,                      1, NT_EVENT,  HID, HID);
        gemm(hEnt, 1, f(B + 17), f(B + 18), qbuf + (size_t)offE * HID, 1, NT_ENTITY, HID, HID);
        gemm(h,    1, f(B + 25), f(B + 26), vbuf,                      1, NT_EVENT,  HID, HID);
        gemm(hEnt, 1, f(B + 23), f(B + 24), vbuf + (size_t)offE * HID, 1, NT_ENTITY, HID, HID);

        // softmax tables + aggregation buffer
        hipMemsetAsync(agg, 0, nH * sizeof(float), stream);
        hipMemsetAsync(den, 0, (size_t)NTOT_N * HEADS * sizeof(float), stream);
        init_amax<<<dim3((NTOT_N * HEADS + 255) / 256), 256, 0, stream>>>(
            amax, (size_t)NTOT_N * HEADS);

        // pass 1: logits + segment max (mentions, next_event, co_occurs)
        edge_attn<<<eblocks(200000), 256, 0, stream>>>(ei_ment, 200000, 0,    offE,
            kbuf, qbuf, f(B + 1), f(B + 15), aE,           amax);
        edge_attn<<<eblocks(150000), 256, 0, stream>>>(ei_next, 150000, 0,    0,
            kbuf, qbuf, f(B + 2), f(B + 16), aE + 800000,  amax);
        edge_attn<<<eblocks(200000), 256, 0, stream>>>(ei_co,   200000, offE, offE,
            kbuf, qbuf, f(B + 0), f(B + 14), aE + 1400000, amax);

        // pass 2: exp + denominator
        edge_exp<<<eblocks(200000), 256, 0, stream>>>(ei_ment, 200000, offE, aE,           amax, den);
        edge_exp<<<eblocks(150000), 256, 0, stream>>>(ei_next, 150000, 0,    aE + 800000,  amax, den);
        edge_exp<<<eblocks(200000), 256, 0, stream>>>(ei_co,   200000, offE, aE + 1400000, amax, den);

        // pass 3: weighted message scatter
        edge_agg<<<eblocks(200000), 256, 0, stream>>>(ei_ment, 200000, 0,    offE,
            vbuf, f(B + 8), aE,           den, agg);
        edge_agg<<<eblocks(150000), 256, 0, stream>>>(ei_next, 150000, 0,    0,
            vbuf, f(B + 9), aE + 800000,  den, agg);
        edge_agg<<<eblocks(200000), 256, 0, stream>>>(ei_co,   200000, offE, offE,
            vbuf, f(B + 7), aE + 1400000, den, agg);

        // gelu(fp32 agg) -> bf16, out_lin GEMM, gated residual + relu (bf16)
        gelu_to_bf<<<dim3((unsigned)((nH + 255) / 256)), 256, 0, stream>>>(agg, aggbf, nH);
        gemm(aggbf,                      1, f(B + 12), f(B + 13), hnew,                      1, NT_EVENT,  HID, HID);
        gemm(aggbf + (size_t)offE * HID, 1, f(B + 10), f(B + 11), hnew + (size_t)offE * HID, 1, NT_ENTITY, HID, HID);

        size_t nEv = (size_t)NT_EVENT * HID, nEn = (size_t)NT_ENTITY * HID;
        residual_relu_bf<<<dim3((unsigned)((nEv + 255) / 256)), 256, 0, stream>>>(
            h, hnew, f(B + 22), nEv);
        residual_relu_bf<<<dim3((unsigned)((nEn + 255) / 256)), 256, 0, stream>>>(
            h + (size_t)offE * HID, hnew + (size_t)offE * HID, f(B + 21), nEn);
    }

    // Output heads (bf16 hidden -> fp32 logits, M=8)
    float* out = (float*)d_out;
    gemm(h,                      1, f(P + 60), f(P + 61), out,                     0, NT_EVENT,  HID, 8);
    gemm(h + (size_t)offE * HID, 1, f(P + 58), f(P + 59), out + (size_t)offE * 8, 0, NT_ENTITY, HID, 8);
}
